// DGMAE_58866821759299
// MI455X (gfx1250) — matrix operand order repaired and checked by reading, not compiled, not run on hardware
//
#include <hip/hip_runtime.h>
#include <math.h>

typedef __attribute__((ext_vector_type(16))) __bf16 v16bf;
typedef __attribute__((ext_vector_type(8)))  float  v8f;

namespace {
constexpr int TT   = 6;
constexpr int NN   = 20000;
constexpr int EE   = 300000;
constexpr int EDD  = 75000;
constexpr int XDIM = 128;
constexpr int HDIM = 256;
constexpr int ZDIM = 128;
constexpr int LDS_STRIDE = 56;   // ushorts per staged column: 112 B, 16B-aligned,
                                 // 28-word bank stride -> conflict-free over 16 lanes
}

// round-to-nearest-even fp32 pair -> packed bf16x2 (lo in [15:0])
static __device__ __forceinline__ unsigned pk2bf(float lo, float hi) {
    unsigned u0 = __builtin_bit_cast(unsigned, lo);
    unsigned u1 = __builtin_bit_cast(unsigned, hi);
    u0 += 0x7FFFu + ((u0 >> 16) & 1u);
    u1 += 0x7FFFu + ((u1 >> 16) & 1u);
    return __builtin_amdgcn_perm(u1, u0, 0x07060302u);  // {u1.hi16, u0.hi16}
}

// ---------------------------------------------------------------------------
// One-time weight prep: fp32 W[K][D] -> bf16 Wt[D][K] (transposed, RNE).
// Contiguous b16 stores; strided loads (weights are tiny / L2-resident).
// ---------------------------------------------------------------------------
__global__ void wt16_kernel(unsigned short* __restrict__ dst,
                            const float* __restrict__ W, int K, int D) {
    int idx = blockIdx.x * blockDim.x + threadIdx.x;
    if (idx >= K * D) return;
    int d = idx / K, k = idx - d * K;
    unsigned u = __builtin_bit_cast(unsigned, W[(size_t)k * D + d]);
    u += 0x7FFFu + ((u >> 16) & 1u);
    dst[idx] = (unsigned short)(u >> 16);
}

// ---------------------------------------------------------------------------
// WMMA GEMM: C[M x Dout] = concat(A1[:,0:K1], A2[:,0:K2]) @ W + bias (+relu).
// Wt is pre-converted bf16 in [Dout][K] layout.  Block = 8 waves = 8 row tiles
// x one 64-col group.  Staging = pure uint4 global->LDS copy (4KB/chunk);
// all four B fragments fetched (8x ds_load_b128) before the 4-WMMA chain.
// Requires M%16==0, Dout%64==0, K1,K2 %32==0.
// ---------------------------------------------------------------------------
__global__ void gemm_wmma_kernel(const float* __restrict__ A1,
                                 const float* __restrict__ A2,
                                 int K1, int K2,
                                 const unsigned short* __restrict__ Wt,
                                 const float* __restrict__ bias,
                                 float* __restrict__ C,
                                 int M, int Dout, int act)
{
    __shared__ alignas(16) unsigned short sW[64 * LDS_STRIDE];

    const int lane = threadIdx.x & 31;
    const int wid  = threadIdx.x >> 5;
    const int rowTiles    = M >> 4;
    const int blocksPerCg = (rowTiles + 7) >> 3;
    const int cgrp = blockIdx.x / blocksPerCg;
    const int brow = blockIdx.x % blocksPerCg;
    const int rt   = brow * 8 + wid;
    const bool active = rt < rowTiles;        // wave-uniform
    const int colBase = cgrp << 6;

    const int m    = lane & 15;               // A row-in-tile == B col-in-tile
    const int half = lane >> 4;
    const int rm   = rt * 16 + m;
    const int K    = K1 + K2;

    const int ccol  = threadIdx.x >> 2;       // staging: column 0..63
    const int cpart = threadIdx.x & 3;        // staging: 16B slice 0..3

    v8f acc[4] = {};

    for (int kc = 0; kc < K; kc += 32) {
        // ---- stage 64 cols x 32 k of bf16 weights: one uint4 per thread ----
        *(uint4*)&sW[ccol * LDS_STRIDE + cpart * 8] =
            *(const uint4*)(Wt + (size_t)(colBase + ccol) * K + kc + cpart * 8);
        __syncthreads();

        if (active) {
            const float* Ap;
            int kb;
            if (kc < K1) { Ap = A1 + (size_t)rm * K1; kb = kc; }
            else         { Ap = A2 + (size_t)rm * K2; kb = kc - K1; }

            // A fragment (ISA 7.12.2): e0..7 -> K=kb+8*half+e ; e8..15 -> +16
            const float4 a0 = *(const float4*)(Ap + kb + 8 * half);
            const float4 a1 = *(const float4*)(Ap + kb + 8 * half + 4);
            const float4 a2 = *(const float4*)(Ap + kb + 16 + 8 * half);
            const float4 a3 = *(const float4*)(Ap + kb + 16 + 8 * half + 4);
            union { v16bf v; unsigned u[8]; } af;
            af.u[0] = pk2bf(a0.x, a0.y); af.u[1] = pk2bf(a0.z, a0.w);
            af.u[2] = pk2bf(a1.x, a1.y); af.u[3] = pk2bf(a1.z, a1.w);
            af.u[4] = pk2bf(a2.x, a2.y); af.u[5] = pk2bf(a2.z, a2.w);
            af.u[6] = pk2bf(a3.x, a3.y); af.u[7] = pk2bf(a3.z, a3.w);

            // fetch ALL four B fragments first so the 8 ds_load_b128 overlap
            union { v16bf v; uint4 q[2]; } bf[4];
            const unsigned short* bp = &sW[m * LDS_STRIDE + 8 * half];
            #pragma unroll
            for (int nt = 0; nt < 4; ++nt) {
                bf[nt].q[0] = *(const uint4*)(bp + nt * 16 * LDS_STRIDE);
                bf[nt].q[1] = *(const uint4*)(bp + nt * 16 * LDS_STRIDE + 16);
            }
            #pragma unroll
            for (int nt = 0; nt < 4; ++nt)
                acc[nt] = __builtin_amdgcn_wmma_f32_16x16x32_bf16(
                    false, af.v, false, bf[nt].v, (short)0, acc[nt], false, false);
        }
        __syncthreads();
    }

    if (active) {
        #pragma unroll
        for (int nt = 0; nt < 4; ++nt) {
            const int col = colBase + (nt << 4) + m;
            const float bv = bias[col];               // never null (zeros buffer)
            #pragma unroll
            for (int v = 0; v < 8; ++v) {
                const int row = rt * 16 + v + 8 * half;  // C/D: M = v + 8*half
                float val = acc[nt][v] + bv;
                if (act == 1) val = fmaxf(val, 0.0f);
                C[(size_t)row * Dout + col] = val;
            }
        }
    }
}

// ---------------------------------------------------------------------------
// GCN aggregation helpers
// ---------------------------------------------------------------------------
__global__ void fill_kernel(float* __restrict__ p, float v, int n) {
    int i = blockIdx.x * blockDim.x + threadIdx.x;
    if (i < n) p[i] = v;
}

__global__ void deg_kernel(float* __restrict__ deg, const int* __restrict__ col, int e) {
    int i = blockIdx.x * blockDim.x + threadIdx.x;
    if (i < e) atomicAdd(&deg[col[i]], 1.0f);
}

__global__ void dinv_kernel(float* __restrict__ dinv, const float* __restrict__ deg, int n) {
    int i = blockIdx.x * blockDim.x + threadIdx.x;
    if (i < n) dinv[i] = rsqrtf(deg[i] + 1.0f);   // +1 for the self-loop
}

__global__ void enorm_kernel(float* __restrict__ enorm, const int* __restrict__ row,
                             const int* __restrict__ col, const float* __restrict__ dinv, int e) {
    int i = blockIdx.x * blockDim.x + threadIdx.x;
    if (i < e) enorm[i] = dinv[row[i]] * dinv[col[i]];
}

// out[i,f] = bias[f] + lin[i,f]*dinv[i]^2   (bias + self-loop term)
__global__ void agg_init_kernel(float* __restrict__ out, const float* __restrict__ lin,
                                const float* __restrict__ bias, const float* __restrict__ dinv,
                                int n, int d) {
    int idx = blockIdx.x * blockDim.x + threadIdx.x;
    if (idx >= n * d) return;
    int i = idx / d, f = idx - i * d;
    float dv = dinv[i];
    out[idx] = bias[f] + lin[idx] * dv * dv;
}

// out[col[e],f] += lin[row[e],f] * enorm[e]
__global__ void agg_edge_kernel(float* __restrict__ out, const float* __restrict__ lin,
                                const int* __restrict__ row, const int* __restrict__ col,
                                const float* __restrict__ enorm, int e, int d) {
    int q4 = d >> 2;
    int gid = blockIdx.x * blockDim.x + threadIdx.x;
    if (gid >= e * q4) return;
    int ei = gid / q4, q = gid - ei * q4;
    float w = enorm[ei];
    const float4 v = *(const float4*)(lin + (size_t)row[ei] * d + 4 * q);
    float* o = out + (size_t)col[ei] * d + 4 * q;
    atomicAdd(o + 0, v.x * w);
    atomicAdd(o + 1, v.y * w);
    atomicAdd(o + 2, v.z * w);
    atomicAdd(o + 3, v.w * w);
}

// ---------------------------------------------------------------------------
// Elementwise kernels
// ---------------------------------------------------------------------------
__global__ void relu_kernel(float* __restrict__ x, int n) {
    int i = blockIdx.x * blockDim.x + threadIdx.x;
    if (i < n) x[i] = fmaxf(x[i], 0.0f);
}
__global__ void softplus_kernel(float* __restrict__ x, int n) {
    int i = blockIdx.x * blockDim.x + threadIdx.x;
    if (i < n) { float v = x[i]; x[i] = (v > 20.0f) ? v : log1pf(expf(v)); }
}
// z_t written both to the (8B-aligned) output slot and an aligned ws copy
__global__ void zt_kernel(float* __restrict__ zo, float* __restrict__ zb,
                          const float* __restrict__ eps, const float* __restrict__ s,
                          const float* __restrict__ m, int n) {
    int i = blockIdx.x * blockDim.x + threadIdx.x;
    if (i < n) { float v = eps[i] * s[i] + m[i]; zo[i] = v; zb[i] = v; }
}
__global__ void sigadd_kernel(float* __restrict__ o, const float* __restrict__ a,
                              const float* __restrict__ b, int n) {
    int i = blockIdx.x * blockDim.x + threadIdx.x;
    if (i < n) o[i] = 1.0f / (1.0f + expf(-(a[i] + b[i])));
}
__global__ void tanhadd_kernel(float* __restrict__ o, const float* __restrict__ a,
                               const float* __restrict__ b, int n) {
    int i = blockIdx.x * blockDim.x + threadIdx.x;
    if (i < n) o[i] = tanhf(a[i] + b[i]);
}
__global__ void mul_kernel(float* __restrict__ o, const float* __restrict__ a,
                           const float* __restrict__ b, int n) {
    int i = blockIdx.x * blockDim.x + threadIdx.x;
    if (i < n) o[i] = a[i] * b[i];
}
__global__ void gru_kernel(float* __restrict__ h, const float* __restrict__ zg,
                           const float* __restrict__ ht, int n) {
    int i = blockIdx.x * blockDim.x + threadIdx.x;
    if (i < n) { float z = zg[i]; h[i] = z * h[i] + (1.0f - z) * ht[i]; }
}
__global__ void copy_kernel(float* __restrict__ d, const float* __restrict__ s, int n) {
    int i = blockIdx.x * blockDim.x + threadIdx.x;
    if (i < n) d[i] = s[i];
}

// ---------------------------------------------------------------------------
// Reductions
// ---------------------------------------------------------------------------
__global__ void kld_kernel(float* __restrict__ acc, const float* __restrict__ m1,
                           const float* __restrict__ s1, const float* __restrict__ m2,
                           const float* __restrict__ s2, int n) {
    int i = blockIdx.x * blockDim.x + threadIdx.x;
    float el = 0.0f;
    if (i < n) {
        float a = s1[i] + 1e-10f, b = s2[i] + 1e-10f, dm = m1[i] - m2[i];
        el = 2.0f * logf(b) - 2.0f * logf(a) + (a * a + dm * dm) / (b * b) - 1.0f;
    }
    #pragma unroll
    for (int off = 16; off; off >>= 1) el += __shfl_xor(el, off, 32);
    if ((threadIdx.x & 31) == 0) atomicAdd(acc, el);
}

// one wave per edge: dot(z[e0], z[e1]) over HDIM, then -log(sig) / -log(1-sig)
__global__ void edge_loss_kernel(float* __restrict__ acc, const float* __restrict__ z,
                                 const int* __restrict__ e0, const int* __restrict__ e1,
                                 int ne, int isNeg) {
    int w = (blockIdx.x * blockDim.x + threadIdx.x) >> 5;
    int lane = threadIdx.x & 31;
    if (w >= ne) return;
    const float* za = z + (size_t)e0[w] * HDIM;
    const float* zb = z + (size_t)e1[w] * HDIM;
    float s = 0.0f;
    #pragma unroll
    for (int j = 0; j < HDIM / 32; ++j) s += za[lane + 32 * j] * zb[lane + 32 * j];
    #pragma unroll
    for (int off = 16; off; off >>= 1) s += __shfl_xor(s, off, 32);
    if (lane == 0) {
        float p = 1.0f / (1.0f + expf(-s));
        float v = isNeg ? -logf(1.0f - p + 1e-10f) : -logf(p + 1e-10f);
        atomicAdd(acc, v);
    }
}

__global__ void finalize_kernel(float* __restrict__ out, const float* __restrict__ acc) {
    out[0] = 0.5f / ((float)NN * (float)NN) * acc[0];
    out[1] = acc[1] / (float)EDD;
}

// ---------------------------------------------------------------------------
// Host orchestration
// ---------------------------------------------------------------------------
extern "C" void kernel_launch(void* const* d_in, const int* in_sizes, int n_in,
                              void* d_out, int out_size, void* d_ws, size_t ws_size,
                              hipStream_t stream) {
    (void)in_sizes; (void)n_in; (void)out_size;
    const float* x       = (const float*)d_in[0];
    const float* phi_x_W = (const float*)d_in[1];  const float* phi_x_b = (const float*)d_in[2];
    const float* phi_z_W = (const float*)d_in[3];  const float* phi_z_b = (const float*)d_in[4];
    const float* enc_W   = (const float*)d_in[5];  const float* enc_b   = (const float*)d_in[6];
    const float* encm_W  = (const float*)d_in[7];  const float* encm_b  = (const float*)d_in[8];
    const float* encs_W  = (const float*)d_in[9];  const float* encs_b  = (const float*)d_in[10];
    const float* prior_W = (const float*)d_in[11]; const float* prior_b = (const float*)d_in[12];
    const float* priorm_W= (const float*)d_in[13]; const float* priorm_b= (const float*)d_in[14];
    const float* priors_W= (const float*)d_in[15]; const float* priors_b= (const float*)d_in[16];
    const float* Wxz = (const float*)d_in[17]; const float* bxz = (const float*)d_in[18];
    const float* Whz = (const float*)d_in[19]; const float* bhz = (const float*)d_in[20];
    const float* Wxr = (const float*)d_in[21]; const float* bxr = (const float*)d_in[22];
    const float* Whr = (const float*)d_in[23]; const float* bhr = (const float*)d_in[24];
    const float* Wxh = (const float*)d_in[25]; const float* bxh = (const float*)d_in[26];
    const float* Whh = (const float*)d_in[27]; const float* bhh = (const float*)d_in[28];
    const float* dec_W = (const float*)d_in[29]; const float* dec_b = (const float*)d_in[30];
    const float* eps1  = (const float*)d_in[31];
    const int* edge_idx     = (const int*)d_in[32];
    const int* edge_dropped = (const int*)d_in[34];
    const int* neg_edges    = (const int*)d_in[35];

    float* out = (float*)d_out;
    const size_t NH = (size_t)NN * HDIM;
    const size_t NZ = (size_t)NN * ZDIM;
    const size_t ENC_OFF = 2;
    const size_t DEC_OFF = ENC_OFF + (size_t)TT * NZ;
    const size_t H_OFF   = DEC_OFF + (size_t)TT * NH;
    const size_t Z_OFF   = H_OFF + NH;

    // workspace carve-up -- every buffer 16B-aligned by construction
    float* ws = (float*)d_ws;
    size_t off = 0;
    auto carve = [&](size_t n) { float* p = ws + off; off += n; return p; };
    float* h       = carve(NH);
    float* phi_x   = carve(NH);
    float* phi_z   = carve(NH);
    float* enc_t   = carve(NH);
    float* prior_t = carve(NH);
    float* lin     = carve(NH);
    float* tmpA    = carve(NH);
    float* tmpB    = carve(NH);
    float* zg      = carve(NH);
    float* rg      = carve(NH);
    float* rh      = carve(NH);
    float* enc_std = carve(NZ);
    float* prior_m = carve(NZ);
    float* prior_s = carve(NZ);
    float* zbuf    = carve(NZ);
    float* deg     = carve(NN);
    float* dinv    = carve(NN);
    float* enorm   = carve(EE);
    float* acc     = carve(8);
    float* zeros   = carve(256);             // branch-free "no bias"
    // bf16 transposed weights (ushort region; base is 16B aligned: all float
    // buffer sizes above are multiples of 8 floats)
    unsigned short* ws16 = (unsigned short*)(ws + off);
    size_t off16 = 0;
    auto carve16 = [&](size_t n) { unsigned short* p = ws16 + off16; off16 += n; return p; };
    unsigned short* wt_phi_x  = carve16((size_t)XDIM * HDIM);
    unsigned short* wt_phi_z  = carve16((size_t)ZDIM * HDIM);
    unsigned short* wt_enc    = carve16((size_t)2 * HDIM * HDIM);
    unsigned short* wt_encm   = carve16((size_t)HDIM * ZDIM);
    unsigned short* wt_encs   = carve16((size_t)HDIM * ZDIM);
    unsigned short* wt_prior  = carve16((size_t)HDIM * HDIM);
    unsigned short* wt_priorm = carve16((size_t)HDIM * ZDIM);
    unsigned short* wt_priors = carve16((size_t)HDIM * ZDIM);
    unsigned short* wt_Wxz    = carve16((size_t)2 * HDIM * HDIM);
    unsigned short* wt_Whz    = carve16((size_t)HDIM * HDIM);
    unsigned short* wt_Wxr    = carve16((size_t)2 * HDIM * HDIM);
    unsigned short* wt_Whr    = carve16((size_t)HDIM * HDIM);
    unsigned short* wt_Wxh    = carve16((size_t)2 * HDIM * HDIM);
    unsigned short* wt_Whh    = carve16((size_t)HDIM * HDIM);
    unsigned short* wt_dec    = carve16((size_t)ZDIM * HDIM);
    if (ws_size < off * sizeof(float) + off16 * sizeof(unsigned short)) return;

    const int THR = 256;
    auto blocks = [](size_t n) { return (int)((n + 255) / 256); };

    auto wcvt = [&](unsigned short* dst, const float* W, int K, int D) {
        wt16_kernel<<<blocks((size_t)K * D), THR, 0, stream>>>(dst, W, K, D);
    };
    auto gemm = [&](const float* A1, const float* A2, int K1, int K2,
                    const unsigned short* Wt, const float* b, float* C, int Dout, int act) {
        int cg = Dout / 64;
        int blocksPerCg = (NN / 16 + 7) / 8;
        gemm_wmma_kernel<<<blocksPerCg * cg, THR, 0, stream>>>(
            A1, A2, K1, K2, Wt, b ? b : zeros, C, NN, Dout, act);
    };
    auto gcn_agg = [&](float* dst, const float* linsrc, const float* bias,
                       const int* rowI, const int* colI, int D) {
        agg_init_kernel<<<blocks((size_t)NN * D), THR, 0, stream>>>(dst, linsrc, bias, dinv, NN, D);
        agg_edge_kernel<<<blocks((size_t)EE * (D / 4)), THR, 0, stream>>>(dst, linsrc, rowI, colI,
                                                                          enorm, EE, D);
    };

    fill_kernel<<<blocks(NH), THR, 0, stream>>>(h, 0.0f, (int)NH);
    fill_kernel<<<1, THR, 0, stream>>>(acc, 0.0f, 8);
    fill_kernel<<<1, THR, 0, stream>>>(zeros, 0.0f, 256);

    // one-time weight transpose + bf16 conversion (shared by all timesteps)
    wcvt(wt_phi_x,  phi_x_W, XDIM, HDIM);
    wcvt(wt_phi_z,  phi_z_W, ZDIM, HDIM);
    wcvt(wt_enc,    enc_W,   2 * HDIM, HDIM);
    wcvt(wt_encm,   encm_W,  HDIM, ZDIM);
    wcvt(wt_encs,   encs_W,  HDIM, ZDIM);
    wcvt(wt_prior,  prior_W, HDIM, HDIM);
    wcvt(wt_priorm, priorm_W,HDIM, ZDIM);
    wcvt(wt_priors, priors_W,HDIM, ZDIM);
    wcvt(wt_Wxz,    Wxz,     2 * HDIM, HDIM);
    wcvt(wt_Whz,    Whz,     HDIM, HDIM);
    wcvt(wt_Wxr,    Wxr,     2 * HDIM, HDIM);
    wcvt(wt_Whr,    Whr,     HDIM, HDIM);
    wcvt(wt_Wxh,    Wxh,     2 * HDIM, HDIM);
    wcvt(wt_Whh,    Whh,     HDIM, HDIM);
    wcvt(wt_dec,    dec_W,   ZDIM, HDIM);

    for (int t = 0; t < TT; ++t) {
        const int* rowI = edge_idx + (size_t)t * 2 * EE;        // edge_index[0] (source)
        const int* colI = rowI + EE;                            // edge_index[1] (target)
        const float* xt = x + (size_t)t * NN * XDIM;
        float* encMean = out + ENC_OFF + (size_t)t * NZ;
        float* zdec    = out + DEC_OFF + (size_t)t * NH;
        float* zt      = out + Z_OFF  + (size_t)t * NZ;

        // GCN normalization for this timestep's edge set
        fill_kernel<<<blocks(NN), THR, 0, stream>>>(deg, 0.0f, NN);
        deg_kernel<<<blocks(EE), THR, 0, stream>>>(deg, colI, EE);
        dinv_kernel<<<blocks(NN), THR, 0, stream>>>(dinv, deg, NN);
        enorm_kernel<<<blocks(EE), THR, 0, stream>>>(enorm, rowI, colI, dinv, EE);

        // phi_x_t = relu(x[t] @ phi_x_W + b)
        gemm(xt, nullptr, XDIM, 0, wt_phi_x, phi_x_b, phi_x, HDIM, 1);

        // enc_t = relu(gcn([phi_x, h], enc_W, enc_b))
        gemm(phi_x, h, HDIM, HDIM, wt_enc, nullptr, lin, HDIM, 0);
        gcn_agg(enc_t, lin, enc_b, rowI, colI, HDIM);
        relu_kernel<<<blocks(NH), THR, 0, stream>>>(enc_t, (int)NH);

        // enc_mean = gcn(enc_t, encm_W, encm_b)  (written directly to output)
        gemm(enc_t, nullptr, HDIM, 0, wt_encm, nullptr, lin, ZDIM, 0);
        gcn_agg(encMean, lin, encm_b, rowI, colI, ZDIM);

        // enc_std = softplus(gcn(enc_t, encs_W, encs_b))
        gemm(enc_t, nullptr, HDIM, 0, wt_encs, nullptr, lin, ZDIM, 0);
        gcn_agg(enc_std, lin, encs_b, rowI, colI, ZDIM);
        softplus_kernel<<<blocks(NZ), THR, 0, stream>>>(enc_std, (int)NZ);

        // prior path (dense)
        gemm(h, nullptr, HDIM, 0, wt_prior, prior_b, prior_t, HDIM, 1);
        gemm(prior_t, nullptr, HDIM, 0, wt_priorm, priorm_b, prior_m, ZDIM, 0);
        gemm(prior_t, nullptr, HDIM, 0, wt_priors, priors_b, prior_s, ZDIM, 0);
        softplus_kernel<<<blocks(NZ), THR, 0, stream>>>(prior_s, (int)NZ);

        // z_t = eps1 * enc_std + enc_mean  (output z_ts[t] + aligned ws copy)
        zt_kernel<<<blocks(NZ), THR, 0, stream>>>(zt, zbuf, eps1, enc_std, encMean, (int)NZ);

        // phi_z_t = relu(z_t @ phi_z_W + b)
        gemm(zbuf, nullptr, ZDIM, 0, wt_phi_z, phi_z_b, phi_z, HDIM, 1);

        // z_dec = gcn(z_t, dec_W, dec_b)  (written directly to output dec_ts[t])
        gemm(zbuf, nullptr, ZDIM, 0, wt_dec, nullptr, lin, HDIM, 0);
        gcn_agg(zdec, lin, dec_b, rowI, colI, HDIM);

        // kld accumulation
        kld_kernel<<<blocks(NZ), THR, 0, stream>>>(acc, encMean, enc_std, prior_m, prior_s, (int)NZ);

        // mask reconstruction loss
        const int* ed = edge_dropped + (size_t)t * 2 * EDD;
        const int* en = neg_edges    + (size_t)t * 2 * EDD;
        int lossBlocks = (int)(((size_t)EDD * 32 + 255) / 256);
        edge_loss_kernel<<<lossBlocks, THR, 0, stream>>>(acc + 1, zdec, ed, ed + EDD, EDD, 0);
        edge_loss_kernel<<<lossBlocks, THR, 0, stream>>>(acc + 1, zdec, en, en + EDD, EDD, 1);

        // graph GRU
        gemm(phi_x, phi_z, HDIM, HDIM, wt_Wxz, nullptr, lin, HDIM, 0);
        gcn_agg(tmpA, lin, bxz, rowI, colI, HDIM);
        gemm(h, nullptr, HDIM, 0, wt_Whz, nullptr, lin, HDIM, 0);
        gcn_agg(tmpB, lin, bhz, rowI, colI, HDIM);
        sigadd_kernel<<<blocks(NH), THR, 0, stream>>>(zg, tmpA, tmpB, (int)NH);

        gemm(phi_x, phi_z, HDIM, HDIM, wt_Wxr, nullptr, lin, HDIM, 0);
        gcn_agg(tmpA, lin, bxr, rowI, colI, HDIM);
        gemm(h, nullptr, HDIM, 0, wt_Whr, nullptr, lin, HDIM, 0);
        gcn_agg(tmpB, lin, bhr, rowI, colI, HDIM);
        sigadd_kernel<<<blocks(NH), THR, 0, stream>>>(rg, tmpA, tmpB, (int)NH);

        gemm(phi_x, phi_z, HDIM, HDIM, wt_Wxh, nullptr, lin, HDIM, 0);
        gcn_agg(tmpA, lin, bxh, rowI, colI, HDIM);
        mul_kernel<<<blocks(NH), THR, 0, stream>>>(rh, rg, h, (int)NH);
        gemm(rh, nullptr, HDIM, 0, wt_Whh, nullptr, lin, HDIM, 0);
        gcn_agg(tmpB, lin, bhh, rowI, colI, HDIM);
        tanhadd_kernel<<<blocks(NH), THR, 0, stream>>>(tmpA, tmpA, tmpB, (int)NH);  // h_t

        gru_kernel<<<blocks(NH), THR, 0, stream>>>(h, zg, tmpA, (int)NH);
    }

    copy_kernel<<<blocks(NH), THR, 0, stream>>>(out + H_OFF, h, (int)NH);
    finalize_kernel<<<1, 1, 0, stream>>>(out, acc);
}
